// MQA_5935644803096
// MI455X (gfx1250) — matrix operand order repaired and checked by reading, not compiled, not run on hardware
//
#include <hip/hip_runtime.h>
#include <math.h>

// ---------------- CDNA5 WMMA plumbing ----------------
typedef __attribute__((ext_vector_type(16))) __bf16 v16bf;
typedef __attribute__((ext_vector_type(8)))  float  v8f;

union Frag { v16bf v; unsigned u[8]; };

__device__ __forceinline__ v8f wmma_bf16(const Frag& a, const Frag& b, v8f c) {
    // D = A(16x32 bf16) * B(32x16 bf16) + C(16x16 f32)
    return __builtin_amdgcn_wmma_f32_16x16x32_bf16(
        false, a.v, false, b.v, (short)0, c, false, false);
}

// ---------------- CDNA5 async global->LDS staging ----------------
// Builtin signature (from compiler diagnostic): param 0 is a pointer to
// int __attribute__((vector_size(16))) in the global address space.
typedef int v4i __attribute__((vector_size(16)));
typedef __attribute__((address_space(1))) v4i g_v4i;   // global
typedef __attribute__((address_space(3))) v4i l_v4i;   // LDS

#if defined(__gfx1250__) && __has_builtin(__builtin_amdgcn_global_load_async_to_lds_b128)
#define HAVE_ASYNC 1
#else
#define HAVE_ASYNC 0
#endif

__device__ __forceinline__ void async_copy_b128(const __bf16* g, __bf16* l) {
#if HAVE_ASYNC
    // low 32 bits of a generic shared pointer are the LDS byte offset
    __builtin_amdgcn_global_load_async_to_lds_b128(
        (g_v4i*)(uintptr_t)g,
        (l_v4i*)(unsigned)(uintptr_t)l,
        0, 0);
#else
    *reinterpret_cast<uint4*>(l) = *reinterpret_cast<const uint4*>(g);
#endif
}

__device__ __forceinline__ void async_fence() {
#if HAVE_ASYNC
#if __has_builtin(__builtin_amdgcn_s_wait_asynccnt)
    __builtin_amdgcn_s_wait_asynccnt(0);
#else
    asm volatile("s_wait_asynccnt 0x0" ::: "memory");
#endif
    asm volatile("" ::: "memory");
#endif
}

// ---------------- constants ----------------
#define BATCH 2
#define SEQ   2048
#define HID   2048
#define HQ    16
#define HKV   2
#define HD    128
#define QKVN  2560              // (16+2*2)*128
#define SCALE_F   0.08838834764831845f
#define SOFTCAP_F 30.0f

// ---------------- fp32 -> bf16 convert (8 elems/thread, packed stores) ----------------
__global__ __launch_bounds__(256) void f2bf_kernel(const float* __restrict__ in,
                                                   __bf16* __restrict__ out, int n8) {
    int i = blockIdx.x * 256 + threadIdx.x;
    if (i >= n8) return;
    const size_t base = (size_t)i * 8;
    float4 a = reinterpret_cast<const float4*>(in + base)[0];
    float4 b = reinterpret_cast<const float4*>(in + base)[1];
    __bf16 h[8];
    h[0] = (__bf16)a.x; h[1] = (__bf16)a.y; h[2] = (__bf16)a.z; h[3] = (__bf16)a.w;
    h[4] = (__bf16)b.x; h[5] = (__bf16)b.y; h[6] = (__bf16)b.z; h[7] = (__bf16)b.w;
    *reinterpret_cast<uint4*>(out + base) = *reinterpret_cast<uint4*>(h);
}

// ---------------- tiled bf16 WMMA GEMM: C[M,N] = A[M,K] * B[K,N] ----------------
// 128x128 block tile, 8 waves in 4(M) x 2(N); each wave 2x4 = 8 WMMA tiles.
// Double-buffered LDS; A tile staged via async global->LDS DMA, B via transpose stores.
__global__ __launch_bounds__(256) void gemm_bf16_kernel(const __bf16* __restrict__ A,
                                                        const __bf16* __restrict__ B,
                                                        float* __restrict__ C,
                                                        int M, int N, int K) {
    __shared__ __bf16 sA[2][128][32];   // [buf][m][k]
    __shared__ __bf16 sBt[2][128][32];  // [buf][n][k] (transposed stage)

    const int tid  = threadIdx.x;
    const int wave = tid >> 5;
    const int lane = tid & 31;
    const int bm = blockIdx.y * 128;
    const int bn = blockIdx.x * 128;
    const int wm = (wave >> 1) * 32;
    const int wn = (wave & 1) * 64;
    const int ln  = lane & 15;
    const int kh8 = (lane >> 4) << 3;   // 0 or 8

    v8f acc[2][4];
    for (int i = 0; i < 2; ++i)
        for (int j = 0; j < 4; ++j)
            for (int e = 0; e < 8; ++e) acc[i][j][e] = 0.0f;

    const int arow = tid >> 1, acol = (tid & 1) * 16;   // A: 128 rows x 32
    const int krow = tid >> 3, bcol = (tid & 7) * 16;   // B: 32 rows x 128

    const __bf16* aptr = A + (size_t)(bm + arow) * K + acol;
    const __bf16* bptr = B + (size_t)krow * N + bn + bcol;

    auto stage = [&](int buf, int k0) {
        // A tile: contiguous rows -> async DMA (2x16B per thread)
        async_copy_b128(aptr + k0,     &sA[buf][arow][acol]);
        async_copy_b128(aptr + k0 + 8, &sA[buf][arow][acol + 8]);
        // B tile: transpose through registers
        const uint4* src = reinterpret_cast<const uint4*>(bptr + (size_t)k0 * N);
        __bf16 tmp[16];
        reinterpret_cast<uint4*>(tmp)[0] = src[0];
        reinterpret_cast<uint4*>(tmp)[1] = src[1];
        for (int e = 0; e < 16; ++e) sBt[buf][bcol + e][krow] = tmp[e];
    };

    int buf = 0;
    stage(0, 0);
    async_fence();
    __syncthreads();

    for (int k0 = 0; k0 < K; k0 += 32) {
        const int nbuf = buf ^ 1;
        if (k0 + 32 < K) stage(nbuf, k0 + 32);

        // fragments (ISA 16-bit A layout: k = e + (e>=8?8:0) + (lane>=16?8:0);
        //            B uses the same k order as A)
        Frag af[2], bfr[4];
        const unsigned* rA0 = reinterpret_cast<const unsigned*>(&sA[buf][wm + ln][0]);
        const unsigned* rA1 = reinterpret_cast<const unsigned*>(&sA[buf][wm + 16 + ln][0]);
        const int b0 = kh8 >> 1;
        for (int j = 0; j < 4; ++j) {
            af[0].u[j] = rA0[b0 + j]; af[0].u[4 + j] = rA0[b0 + 8 + j];
            af[1].u[j] = rA1[b0 + j]; af[1].u[4 + j] = rA1[b0 + 8 + j];
        }
        for (int jj = 0; jj < 4; ++jj) {
            const unsigned* rB = reinterpret_cast<const unsigned*>(&sBt[buf][wn + 16 * jj + ln][0]);
            for (int e = 0; e < 8; ++e) bfr[jj].u[e] = rB[(kh8 >> 1) + (e & 3) + ((e >> 2) << 3)];
        }
        for (int jj = 0; jj < 4; ++jj)
            for (int i = 0; i < 2; ++i)
                acc[i][jj] = wmma_bf16(af[i], bfr[jj], acc[i][jj]);

        async_fence();
        __syncthreads();
        buf = nbuf;
    }

    // store (C layout: M = r + 8*(lane>=16), N = lane&15)
    for (int i = 0; i < 2; ++i)
        for (int j = 0; j < 4; ++j) {
            const int crow0 = bm + wm + 16 * i + kh8;
            const int ccol  = bn + wn + 16 * j + ln;
            for (int r = 0; r < 8; ++r)
                C[(size_t)(crow0 + r) * N + ccol] = acc[i][j][r];
        }
}

// ---------------- RoPE + split heads: qkv fp32 -> Q/K/V bf16 [B,H,S,D] ----------------
__global__ __launch_bounds__(256) void rope_split_kernel(const float* __restrict__ qkv,
                                                         __bf16* __restrict__ Qb,
                                                         __bf16* __restrict__ Kb,
                                                         __bf16* __restrict__ Vb) {
    const int total = BATCH * SEQ * (HQ + 2 * HKV) * (HD / 2);
    int idx = blockIdx.x * 256 + threadIdx.x;
    if (idx >= total) return;
    const int p  = idx & 63;
    const int hh = (idx >> 6) % (HQ + 2 * HKV);
    const int t  = idx / (64 * (HQ + 2 * HKV));
    const int b = t / SEQ, s = t % SEQ;

    const float* src = qkv + (size_t)t * QKVN + hh * HD;
    float x1 = src[p], x2 = src[p + 64];

    if (hh < HQ + HKV) {
        const float freq = (float)s * __powf(10000.0f, -(float)(2 * p) / (float)HD);
        float sn, cs;
        __sincosf(freq, &sn, &cs);
        const float o1 = x1 * cs - x2 * sn;
        const float o2 = x1 * sn + x2 * cs;
        x1 = o1; x2 = o2;
    }
    __bf16* dst;
    if (hh < HQ)
        dst = Qb + (((size_t)b * HQ + hh) * SEQ + s) * HD;
    else if (hh < HQ + HKV)
        dst = Kb + (((size_t)b * HKV + (hh - HQ)) * SEQ + s) * HD;
    else
        dst = Vb + (((size_t)b * HKV + (hh - HQ - HKV)) * SEQ + s) * HD;
    dst[p]      = (__bf16)x1;
    dst[p + 64] = (__bf16)x2;
}

// ---------------- flash attention (causal, softcap), bf16 WMMA ----------------
// grid: (S/128, HQ, B); block: 256 threads (8 waves). Wave w owns q rows [q0+16w, q0+16w+16).
__global__ __launch_bounds__(256) void attn_kernel(const __bf16* __restrict__ Qb,
                                                   const __bf16* __restrict__ Kb,
                                                   const __bf16* __restrict__ Vb,
                                                   __bf16* __restrict__ Ob) {
    __shared__ __bf16 sK[64][128];    // [k][d]   16 KB (async staged)
    __shared__ __bf16 sVt[128][64];   // [d][k]   16 KB (transposed)
    __shared__ __bf16 sP[8][16][64];  // per-wave P scratch, 16 KB

    const int b  = blockIdx.z;
    const int h  = blockIdx.y;
    const int q0 = blockIdx.x * 128;
    const int kvh = h >> 3;           // GQA: 16 q heads / 2 kv heads

    const int tid  = threadIdx.x;
    const int wave = tid >> 5;
    const int lane = tid & 31;
    const int ln   = lane & 15;
    const int kh8  = (lane >> 4) << 3;
    const int qw   = q0 + wave * 16;

    const __bf16* Qbase = Qb + (((size_t)b * HQ  + h  ) * SEQ) * HD;
    const __bf16* Kbase = Kb + (((size_t)b * HKV + kvh) * SEQ) * HD;
    const __bf16* Vbase = Vb + (((size_t)b * HKV + kvh) * SEQ) * HD;

    // load Q fragments for this wave's 16 rows (4 chunks of d=32)
    Frag qa[4];
    {
        const unsigned* qrow = reinterpret_cast<const unsigned*>(Qbase + (size_t)(qw + ln) * HD);
        for (int c = 0; c < 4; ++c) {
            const int b0 = 16 * c + (kh8 >> 1);
            for (int j = 0; j < 4; ++j) {
                qa[c].u[j]     = qrow[b0 + j];
                qa[c].u[4 + j] = qrow[b0 + 8 + j];
            }
        }
    }

    v8f o[8];
    for (int dt = 0; dt < 8; ++dt)
        for (int e = 0; e < 8; ++e) o[dt][e] = 0.0f;
    float mrow[8], lrow[8];
    for (int r = 0; r < 8; ++r) { mrow[r] = -1e30f; lrow[r] = 0.0f; }

    const int srow = tid >> 2, sc0 = (tid & 3) * 32;   // staging map: 64 rows x 128
    const int kend = q0 + 128;                          // causal bound for this q tile
    for (int kb = 0; kb < kend; kb += 64) {
        // stage K block [64][128] via async DMA (4x16B per thread)
        {
            const __bf16* g = Kbase + (size_t)(kb + srow) * HD + sc0;
            async_copy_b128(g,      &sK[srow][sc0]);
            async_copy_b128(g + 8,  &sK[srow][sc0 + 8]);
            async_copy_b128(g + 16, &sK[srow][sc0 + 16]);
            async_copy_b128(g + 24, &sK[srow][sc0 + 24]);
        }
        // stage V block transposed [128][64]
        {
            const uint4* src = reinterpret_cast<const uint4*>(Vbase + (size_t)(kb + srow) * HD + sc0);
            __bf16 tmp[32];
            reinterpret_cast<uint4*>(tmp)[0] = src[0];
            reinterpret_cast<uint4*>(tmp)[1] = src[1];
            reinterpret_cast<uint4*>(tmp)[2] = src[2];
            reinterpret_cast<uint4*>(tmp)[3] = src[3];
            for (int e = 0; e < 32; ++e) sVt[sc0 + e][srow] = tmp[e];
        }
        async_fence();
        __syncthreads();

        if (kb <= qw + 15) {   // uniform per wave: skip fully-masked k blocks
            // ---- S = Q K^T over 4 column tiles of 16 ----
            float st[4][8];
            for (int jt = 0; jt < 4; ++jt) {
                v8f s;
                for (int e = 0; e < 8; ++e) s[e] = 0.0f;
                const unsigned* rowK = reinterpret_cast<const unsigned*>(&sK[16 * jt + ln][0]);
                for (int c = 0; c < 4; ++c) {
                    Frag kf;
                    for (int e = 0; e < 8; ++e) kf.u[e] = rowK[16 * c + (kh8 >> 1) + (e & 3) + ((e >> 2) << 3)];
                    s = wmma_bf16(qa[c], kf, s);
                }
                for (int r = 0; r < 8; ++r) {
                    float z = s[r] * SCALE_F;
                    z = SOFTCAP_F * tanhf(z * (1.0f / SOFTCAP_F));
                    const int row = qw + r + kh8;
                    const int col = kb + 16 * jt + ln;
                    st[jt][r] = (col <= row) ? z : -1e30f;
                }
            }
            // ---- online softmax ----
            float alpha[8];
            for (int r = 0; r < 8; ++r) {
                float mx = st[0][r];
                for (int jt = 1; jt < 4; ++jt) mx = fmaxf(mx, st[jt][r]);
                for (int off = 1; off < 16; off <<= 1)
                    mx = fmaxf(mx, __shfl_xor(mx, off, 32));
                const float mn = fmaxf(mrow[r], mx);
                alpha[r] = __expf(mrow[r] - mn);
                mrow[r] = mn;
            }
            float rsum[8];
            for (int r = 0; r < 8; ++r) rsum[r] = 0.0f;
            for (int jt = 0; jt < 4; ++jt)
                for (int r = 0; r < 8; ++r) {
                    const float p = __expf(st[jt][r] - mrow[r]);
                    st[jt][r] = p;
                    rsum[r] += p;
                }
            for (int r = 0; r < 8; ++r) {
                float rs = rsum[r];
                for (int off = 1; off < 16; off <<= 1)
                    rs += __shfl_xor(rs, off, 32);
                lrow[r] = lrow[r] * alpha[r] + rs;
                for (int dt = 0; dt < 8; ++dt) o[dt][r] *= alpha[r];
            }
            // ---- C-layout -> A-layout via per-wave LDS scratch ----
            for (int jt = 0; jt < 4; ++jt)
                for (int r = 0; r < 8; ++r)
                    sP[wave][r + kh8][16 * jt + ln] = (__bf16)st[jt][r];
            // ---- O += P V ----
            const unsigned* prow = reinterpret_cast<const unsigned*>(&sP[wave][ln][0]);
            for (int ka = 0; ka < 2; ++ka) {
                Frag pa;
                const int pb = 16 * ka + (kh8 >> 1);
                for (int j = 0; j < 4; ++j) {
                    pa.u[j]     = prow[pb + j];
                    pa.u[4 + j] = prow[pb + 8 + j];
                }
                for (int dt = 0; dt < 8; ++dt) {
                    Frag vf;
                    const unsigned* rowV = reinterpret_cast<const unsigned*>(&sVt[16 * dt + ln][0]);
                    for (int e = 0; e < 8; ++e) vf.u[e] = rowV[16 * ka + (kh8 >> 1) + (e & 3) + ((e >> 2) << 3)];
                    o[dt] = wmma_bf16(pa, vf, o[dt]);
                }
            }
        }
        __syncthreads();
    }

    // epilogue: normalize and store bf16 [token][h*128+d]
    for (int r = 0; r < 8; ++r) {
        const float inv = 1.0f / lrow[r];
        const int row = qw + r + kh8;
        __bf16* dst = Ob + ((size_t)(b * SEQ + row)) * (HQ * HD) + h * HD;
        for (int dt = 0; dt < 8; ++dt)
            dst[16 * dt + ln] = (__bf16)(o[dt][r] * inv);
    }
}

// ---------------- host launcher ----------------
extern "C" void kernel_launch(void* const* d_in, const int* in_sizes, int n_in,
                              void* d_out, int out_size, void* d_ws, size_t ws_size,
                              hipStream_t stream) {
    const float* hidden = (const float*)d_in[0];   // [B,S,HID] fp32
    const float* Wqkv   = (const float*)d_in[1];   // [HID,QKVN] fp32
    const float* Wo     = (const float*)d_in[2];   // [HQ*HD,HID] fp32
    float* out = (float*)d_out;                    // [B,S,HID] fp32

    const size_t TOK = (size_t)BATCH * SEQ;        // 4096

    char* ws = (char*)d_ws;
    size_t off = 0;
    auto alloc = [&](size_t bytes) -> void* {
        void* p = ws + off;
        off += (bytes + 255) & ~(size_t)255;
        return p;
    };
    __bf16* hb    = (__bf16*)alloc(TOK * HID * sizeof(__bf16));
    __bf16* wqkvb = (__bf16*)alloc((size_t)HID * QKVN * sizeof(__bf16));
    __bf16* wob   = (__bf16*)alloc((size_t)HID * HID * sizeof(__bf16));
    float*  qkvf  = (float*) alloc(TOK * QKVN * sizeof(float));
    __bf16* Qb    = (__bf16*)alloc((size_t)BATCH * HQ  * SEQ * HD * sizeof(__bf16));
    __bf16* Kb    = (__bf16*)alloc((size_t)BATCH * HKV * SEQ * HD * sizeof(__bf16));
    __bf16* Vb    = (__bf16*)alloc((size_t)BATCH * HKV * SEQ * HD * sizeof(__bf16));
    __bf16* attnb = (__bf16*)alloc(TOK * (size_t)(HQ * HD) * sizeof(__bf16));
    (void)ws_size;

    // 1) fp32 -> bf16 conversions (8 elems/thread)
    {
        int n0 = (int)(TOK * HID / 8);
        f2bf_kernel<<<dim3((n0 + 255) / 256), dim3(256), 0, stream>>>(hidden, hb, n0);
        int n1 = HID * QKVN / 8;
        f2bf_kernel<<<dim3((n1 + 255) / 256), dim3(256), 0, stream>>>(Wqkv, wqkvb, n1);
        int n2 = HID * HID / 8;
        f2bf_kernel<<<dim3((n2 + 255) / 256), dim3(256), 0, stream>>>(Wo, wob, n2);
    }
    // 2) QKV GEMM: [4096,2048] x [2048,2560] -> fp32
    gemm_bf16_kernel<<<dim3(QKVN / 128, TOK / 128), dim3(256), 0, stream>>>(
        hb, wqkvb, qkvf, (int)TOK, QKVN, HID);
    // 3) RoPE + head split (bf16)
    {
        int n = BATCH * SEQ * (HQ + 2 * HKV) * (HD / 2);
        rope_split_kernel<<<dim3((n + 255) / 256), dim3(256), 0, stream>>>(qkvf, Qb, Kb, Vb);
    }
    // 4) flash attention
    attn_kernel<<<dim3(SEQ / 128, HQ, BATCH), dim3(256), 0, stream>>>(Qb, Kb, Vb, attnb);
    // 5) output GEMM: [4096,2048] x [2048,2048] -> fp32 d_out
    gemm_bf16_kernel<<<dim3(HID / 128, TOK / 128), dim3(256), 0, stream>>>(
        attnb, wob, out, (int)TOK, HID, HID);
}
